// RoPE_Attention_35270271435673
// MI455X (gfx1250) — matrix operand order repaired and checked by reading, not compiled, not run on hardware
//
#include <hip/hip_runtime.h>
#include <hip/hip_bf16.h>

// ---------------------------------------------------------------------------
// RoPE attention block for MI455X (gfx1250, wave32, WMMA bf16 16x16x32).
//   1) convert x / qkv_w / proj_w fp32 -> bf16 (one pass, HBM-cheap)
//   2) fused QKV GEMM + bias + RoPE -> q/k/v in [H][N][128] bf16
//      (double-buffered async global->LDS staging, ASYNCcnt-pipelined)
//   3) flash attention per (head, segment, 128-query block), f32 softmax
//   4) proj GEMM -> fp32 output + bias
// ---------------------------------------------------------------------------

typedef __attribute__((ext_vector_type(16))) __bf16 v16bf;
typedef __attribute__((ext_vector_type(8)))  __bf16 v8bf;
typedef __attribute__((ext_vector_type(8)))  float  v8f;
typedef int v4i32 __attribute__((vector_size(16)));  // matches builtin proto

#if defined(__AMDGCN__) &&                                            \
    __has_builtin(__builtin_amdgcn_global_load_async_to_lds_b128) && \
    __has_builtin(__builtin_amdgcn_s_wait_asynccnt)
#define USE_ASYNC_LDS 1
#else
#define USE_ASYNC_LDS 0
#endif

__device__ __forceinline__ v8f v8f_zero() {
  v8f r;
#pragma unroll
  for (int i = 0; i < 8; ++i) r[i] = 0.0f;
  return r;
}

// A-matrix fragment (16-bit, 16x32): lane L holds row M=L%16; elements 0..15 are
// K = (L>=16?16:0)+e (contiguous, matches B fragments). Two 16B loads.
__device__ __forceinline__ v16bf load_frag_a(const __bf16* p0, const __bf16* p1) {
  v8bf lo = *(const v8bf*)p0;
  v8bf hi = *(const v8bf*)p1;
  v16bf r;
#pragma unroll
  for (int i = 0; i < 8; ++i) { r[i] = lo[i]; r[i + 8] = hi[i]; }
  return r;
}

// B-matrix fragment (16-bit, 32x16): lane L holds column N=L%16; lanes 0-15
// hold K=0..15, lanes 16-31 hold K=16..31 -> 16 contiguous elements (32B).
__device__ __forceinline__ v16bf load_frag_c(const __bf16* p) {
  v8bf lo = *(const v8bf*)p;
  v8bf hi = *(const v8bf*)(p + 8);
  v16bf r;
#pragma unroll
  for (int i = 0; i < 8; ++i) { r[i] = lo[i]; r[i + 8] = hi[i]; }
  return r;
}

// Stage one 16-bf16 (32B) strip global -> LDS. Async path issues two B128
// async-to-LDS ops (INST_OFFSET applies to both global and LDS addresses per
// ISA 10.x pseudocode); fallback path round-trips through VGPRs.
__device__ __forceinline__ void stage32B(const __bf16* g, __bf16* l) {
#if USE_ASYNC_LDS
  __attribute__((address_space(1))) v4i32* gp =
      (__attribute__((address_space(1))) v4i32*)g;
  __attribute__((address_space(3))) v4i32* lp =
      (__attribute__((address_space(3))) v4i32*)l;
  __builtin_amdgcn_global_load_async_to_lds_b128(gp, lp, 0, 0);
  __builtin_amdgcn_global_load_async_to_lds_b128(gp, lp, 16, 0);
#else
  const uint4* gs = (const uint4*)g;
  uint4 a0 = gs[0], a1 = gs[1];
  uint4* ld = (uint4*)l;
  ld[0] = a0;
  ld[1] = a1;
#endif
}

__device__ __forceinline__ void wait_async_le4() {
#if USE_ASYNC_LDS
  __builtin_amdgcn_s_wait_asynccnt(4);
#endif
}
__device__ __forceinline__ void wait_async_le0() {
#if USE_ASYNC_LDS
  __builtin_amdgcn_s_wait_asynccnt(0);
#endif
}

// ---------------------------------------------------------------------------
// fp32 -> bf16 conversion
// ---------------------------------------------------------------------------
__global__ void f32_to_bf16(const float* __restrict__ src,
                            __bf16* __restrict__ dst, int n) {
  int i = blockIdx.x * blockDim.x + threadIdx.x;
  int stride = gridDim.x * blockDim.x;
  for (; i < n; i += stride) dst[i] = (__bf16)src[i];
}

// ---------------------------------------------------------------------------
// Tiled GEMM: C[m][n] = sum_k A[m][k] * B[n][k] (+bias)
// Block tile 128x128, 8 waves, each wave 32x64 (2 M-tiles x 4 N-tiles).
// Double-buffered LDS panels with async global->LDS staging: panel i+1 is in
// flight (ASYNCcnt) while WMMAs consume panel i.
// MODE 0: fp32 output + bias (projection)
// MODE 1: bias + RoPE epilogue, scatter q/k/v bf16 into [H][N][128]
// ---------------------------------------------------------------------------
template <int MODE>
__global__ __launch_bounds__(256, 2)
void gemm128(const __bf16* __restrict__ A, const __bf16* __restrict__ B,
             const float* __restrict__ bias,
             const float* __restrict__ fcos, const float* __restrict__ fsin,
             __bf16* __restrict__ qo, __bf16* __restrict__ ko,
             __bf16* __restrict__ vo, float* __restrict__ co,
             int M, int Ncols, int K) {
  __shared__ __bf16 As[2][128 * 32];
  __shared__ __bf16 Bs[2][128 * 32];

  const int tid  = threadIdx.x;
  const int wave = tid >> 5;
  const int lane = tid & 31;
  const int l16  = lane & 15;
  const int hh   = lane >> 4;  // half-wave select

  const int nb = Ncols >> 7;
  const int m0 = (blockIdx.x / nb) << 7;
  const int n0 = (blockIdx.x % nb) << 7;
  const int wm = (wave >> 1) * 32;  // 4 waves along M
  const int wn = (wave & 1) * 64;   // 2 waves along N

  v8f acc[2][4];
#pragma unroll
  for (int i = 0; i < 2; ++i)
#pragma unroll
    for (int j = 0; j < 4; ++j) acc[i][j] = v8f_zero();

  const int srow = tid >> 1;        // 0..127
  const int sko  = (tid & 1) << 4;  // 0 or 16 elements

  const __bf16* agp = A + (size_t)(m0 + srow) * K + sko;
  const __bf16* bgp = B + (size_t)(n0 + srow) * K + sko;
  __bf16* alp0 = &As[0][srow * 32 + sko];
  __bf16* blp0 = &Bs[0][srow * 32 + sko];
  __bf16* alp1 = &As[1][srow * 32 + sko];
  __bf16* blp1 = &Bs[1][srow * 32 + sko];

  const int nsteps = K >> 5;

  // prologue: stage panel 0 (4 async B128 ops per thread)
  stage32B(agp, alp0);
  stage32B(bgp, blp0);

  for (int i = 0; i < nsteps; ++i) {
    const int cur = i & 1;
    if (i + 1 < nsteps) {
      // stage panel i+1 into the other buffer while panel i is consumed
      const int kk = (i + 1) << 5;
      stage32B(agp + kk, cur ? alp0 : alp1);
      stage32B(bgp + kk, cur ? blp0 : blp1);
      wait_async_le4();  // panel i complete, panel i+1 still in flight
    } else {
      wait_async_le0();
    }
    __syncthreads();

    v16bf af[2], bfr[4];
#pragma unroll
    for (int mt = 0; mt < 2; ++mt) {
      const __bf16* p = &As[cur][(wm + mt * 16 + l16) * 32 + hh * 16];
      af[mt] = load_frag_a(p, p + 8);
    }
#pragma unroll
    for (int nt = 0; nt < 4; ++nt)
      bfr[nt] = load_frag_c(&Bs[cur][(wn + nt * 16 + l16) * 32 + hh * 16]);

#pragma unroll
    for (int mt = 0; mt < 2; ++mt)
#pragma unroll
      for (int nt = 0; nt < 4; ++nt)
        acc[mt][nt] = __builtin_amdgcn_wmma_f32_16x16x32_bf16(
            false, af[mt], false, bfr[nt], (short)0, acc[mt][nt], false, false);

    __syncthreads();  // all waves done reading buf[cur] before it is restaged
  }

  // epilogue (C/D layout: VGPR v, lanes 0-15 -> M=v, lanes 16-31 -> M=v+8)
#pragma unroll
  for (int mt = 0; mt < 2; ++mt) {
#pragma unroll
    for (int nt = 0; nt < 4; ++nt) {
      const int col = n0 + wn + nt * 16 + l16;
      if (MODE == 0) {
        const float b = bias[col];
#pragma unroll
        for (int v = 0; v < 8; ++v) {
          const int row = m0 + wm + mt * 16 + hh * 8 + v;
          co[(size_t)row * Ncols + col] = acc[mt][nt][v] + b;
        }
      } else {
        const int which = col >> 11;       // 0=q, 1=k, 2=v
        const int head  = (col >> 7) & 15;
        const int d     = col & 127;
        const float b   = bias[col];
#pragma unroll
        for (int v = 0; v < 8; ++v) {
          const int row = m0 + wm + mt * 16 + hh * 8 + v;
          float val = acc[mt][nt][v] + b;
          // RoPE pairs (2i, 2i+1) are adjacent lanes -> xor-1 shuffle
          float partner = __shfl_xor(val, 1);
          if (which == 2) {
            vo[((size_t)head * M + row) * 128 + d] = (__bf16)val;
          } else {
            const int fi  = d >> 1;
            const float c = fcos[(size_t)row * 64 + fi];
            const float s = fsin[(size_t)row * 64 + fi];
            // even d holds real part, odd d holds imag part
            float o = (d & 1) ? (partner * s + val * c)
                              : (val * c - partner * s);
            __bf16* dst = (which == 0) ? qo : ko;
            dst[((size_t)head * M + row) * 128 + d] = (__bf16)o;
          }
        }
      }
    }
  }
}

// ---------------------------------------------------------------------------
// Flash attention: block = (head, segment, 128-query block), 8 waves,
// each wave owns 16 query rows. Keys/values streamed in chunks of 32 via LDS.
// ---------------------------------------------------------------------------
__global__ __launch_bounds__(256, 2)
void attn_fa(const __bf16* __restrict__ qg, const __bf16* __restrict__ kg,
             const __bf16* __restrict__ vg, const int* __restrict__ cu,
             __bf16* __restrict__ ao, int Ntok) {
  __shared__ __bf16 Ks[32 * 128];    // keys, row-major [key][d]
  __shared__ __bf16 VT[128 * 32];    // values transposed [d][key]
  __shared__ __bf16 Ps[8][16 * 32];  // per-wave P tile (C-layout -> A-layout)

  const int tid  = threadIdx.x;
  const int wave = tid >> 5;
  const int lane = tid & 31;
  const int l16  = lane & 15;
  const int hh   = lane >> 4;

  const int bpseg = 6;  // 768 / 128
  const int head = blockIdx.x / (4 * bpseg);
  const int rem  = blockIdx.x % (4 * bpseg);
  const int seg  = rem / bpseg;
  const int qb   = rem % bpseg;
  const int lo   = cu[seg];
  const int hi   = cu[seg + 1];
  const int q0   = lo + qb * 128 + wave * 16;

  // Q tile resident in A-fragments: 16 rows x 128 d = 4 K-steps of 32
  v16bf qf[4];
  {
    const __bf16* qp = qg + ((size_t)head * Ntok + q0 + l16) * 128 + hh * 16;
#pragma unroll
    for (int dc = 0; dc < 4; ++dc)
      qf[dc] = load_frag_a(qp + dc * 32, qp + dc * 32 + 8);
  }

  v8f o[8];  // 16 rows x 128 d output accumulator (8 N-tiles)
#pragma unroll
  for (int t = 0; t < 8; ++t) o[t] = v8f_zero();
  float mrow[8], lrow[8];
#pragma unroll
  for (int v = 0; v < 8; ++v) { mrow[v] = -1e30f; lrow[v] = 0.0f; }

  const int skey = tid >> 3;        // 0..31
  const int sdp  = (tid & 7) << 4;  // 0,16,...,112

  for (int kc = lo; kc < hi; kc += 32) {
    __syncthreads();
    {
      // stage K (row-major) and V (transposed) chunk: 32 keys x 128 d
      const uint4* ks = (const uint4*)(kg + ((size_t)head * Ntok + kc + skey) * 128 + sdp);
      uint4 k0 = ks[0], k1 = ks[1];
      uint4* kd = (uint4*)(Ks + skey * 128 + sdp);
      kd[0] = k0; kd[1] = k1;
      const __bf16* vs = vg + ((size_t)head * Ntok + kc + skey) * 128 + sdp;
      v8bf v0 = *(const v8bf*)vs;
      v8bf v1 = *(const v8bf*)(vs + 8);
#pragma unroll
      for (int i = 0; i < 8; ++i) {
        VT[(sdp + i) * 32 + skey]     = v0[i];
        VT[(sdp + 8 + i) * 32 + skey] = v1[i];
      }
    }
    __syncthreads();

    // S = Q K^T for two 16-key column tiles
    v8f s0 = v8f_zero(), s1 = v8f_zero();
#pragma unroll
    for (int dc = 0; dc < 4; ++dc) {
      v16bf b0 = load_frag_c(Ks + (0 + l16) * 128 + dc * 32 + hh * 16);
      v16bf b1 = load_frag_c(Ks + (16 + l16) * 128 + dc * 32 + hh * 16);
      s0 = __builtin_amdgcn_wmma_f32_16x16x32_bf16(false, qf[dc], false, b0,
                                                   (short)0, s0, false, false);
      s1 = __builtin_amdgcn_wmma_f32_16x16x32_bf16(false, qf[dc], false, b1,
                                                   (short)0, s1, false, false);
    }

    // online softmax (row = v + 8*hh, columns across the 16 lanes of a half)
    const float scale = 0.08838834764831845f;  // 1/sqrt(128)
#pragma unroll
    for (int v = 0; v < 8; ++v) {
      float a = s0[v] * scale;
      float b = s1[v] * scale;
      float mx = fmaxf(a, b);
#pragma unroll
      for (int off = 8; off >= 1; off >>= 1)
        mx = fmaxf(mx, __shfl_xor(mx, off));
      const float mn    = fmaxf(mrow[v], mx);
      const float alpha = __expf(mrow[v] - mn);
      mrow[v] = mn;
      a = __expf(a - mn);
      b = __expf(b - mn);
      float rs = a + b;
#pragma unroll
      for (int off = 8; off >= 1; off >>= 1)
        rs += __shfl_xor(rs, off);
      lrow[v] = lrow[v] * alpha + rs;
#pragma unroll
      for (int t = 0; t < 8; ++t) o[t][v] *= alpha;
      const int prow = v + hh * 8;
      Ps[wave][prow * 32 + l16]      = (__bf16)a;
      Ps[wave][prow * 32 + 16 + l16] = (__bf16)b;
    }

    // O += P @ V  (P re-read from per-wave LDS in A-layout)
    {
      const __bf16* pp = &Ps[wave][l16 * 32 + hh * 16];
      v16bf pf = load_frag_a(pp, pp + 8);
#pragma unroll
      for (int t = 0; t < 8; ++t) {
        v16bf vf = load_frag_c(VT + (t * 16 + l16) * 32 + hh * 16);
        o[t] = __builtin_amdgcn_wmma_f32_16x16x32_bf16(false, pf, false, vf,
                                                       (short)0, o[t], false, false);
      }
    }
  }

  // normalize and scatter to [N][C] bf16 (input of the projection GEMM)
  float inv[8];
#pragma unroll
  for (int v = 0; v < 8; ++v) inv[v] = 1.0f / lrow[v];
#pragma unroll
  for (int t = 0; t < 8; ++t) {
#pragma unroll
    for (int v = 0; v < 8; ++v) {
      const int row = q0 + hh * 8 + v;
      const int col = head * 128 + t * 16 + l16;
      ao[(size_t)row * 2048 + col] = (__bf16)(o[t][v] * inv[v]);
    }
  }
}

// ---------------------------------------------------------------------------
// launcher
// ---------------------------------------------------------------------------
extern "C" void kernel_launch(void* const* d_in, const int* in_sizes, int n_in,
                              void* d_out, int out_size, void* d_ws,
                              size_t ws_size, hipStream_t stream) {
  const float* x      = (const float*)d_in[0];
  const float* fcos   = (const float*)d_in[1];
  const float* fsin   = (const float*)d_in[2];
  const float* qkv_w  = (const float*)d_in[3];
  const float* qkv_b  = (const float*)d_in[4];
  const float* proj_w = (const float*)d_in[5];
  const float* proj_b = (const float*)d_in[6];
  const int*   cu     = (const int*)d_in[7];

  const int N = 3072, C = 2048, H = 16, HD = 128;
  const int NQ = 3 * C;  // 6144

  char* w = (char*)d_ws;
  __bf16* x_bf    = (__bf16*)w; w += (size_t)N * C * 2;
  __bf16* qw_bf   = (__bf16*)w; w += (size_t)NQ * C * 2;
  __bf16* pw_bf   = (__bf16*)w; w += (size_t)C * C * 2;
  __bf16* q_ws    = (__bf16*)w; w += (size_t)H * N * HD * 2;
  __bf16* k_ws    = (__bf16*)w; w += (size_t)H * N * HD * 2;
  __bf16* v_ws    = (__bf16*)w; w += (size_t)H * N * HD * 2;
  __bf16* attn_bf = (__bf16*)w; w += (size_t)N * C * 2;

  f32_to_bf16<<<512, 256, 0, stream>>>(x, x_bf, N * C);
  f32_to_bf16<<<1024, 256, 0, stream>>>(qkv_w, qw_bf, NQ * C);
  f32_to_bf16<<<512, 256, 0, stream>>>(proj_w, pw_bf, C * C);

  // fused QKV GEMM + bias + RoPE -> head-major q/k/v
  gemm128<1><<<(N / 128) * (NQ / 128), 256, 0, stream>>>(
      x_bf, qw_bf, qkv_b, fcos, fsin, q_ws, k_ws, v_ws, nullptr, N, NQ, C);

  // block-diagonal flash attention (segments from cu_seqlens)
  attn_fa<<<H * 4 * 6, 256, 0, stream>>>(q_ws, k_ws, v_ws, cu, attn_bf, N);

  // output projection, fp32 + bias
  gemm128<0><<<(N / 128) * (C / 128), 256, 0, stream>>>(
      attn_bf, pw_bf, proj_b, nullptr, nullptr, nullptr, nullptr, nullptr,
      (float*)d_out, N, C, C);
}
